// PairformerModule_14147622273790
// MI455X (gfx1250) — compile-verified, matrix operand order checked by reading, not run on hardware
//
#include <hip/hip_runtime.h>
#include <hip/hip_bf16.h>
#include <cmath>

// ---------------------------------------------------------------------------
// Pairformer (AF3-style) forward for MI455X / gfx1250.
// All matmuls: batched, fully-strided bf16-WMMA GEMM (v_wmma_f32_16x16x32_bf16,
// f32 accumulate), 32x32 C-tile per wave (2x2 fragments, 4 WMMA per K-step),
// native bf16 conversion, b128 vector fragment loads on contiguous-K paths.
// Working set (~165MB) is L2-resident (192MB L2) -> compute-bound on WMMA.
// ---------------------------------------------------------------------------

typedef __bf16 bf16_t;
typedef bf16_t v16bf __attribute__((ext_vector_type(16)));
typedef float  v8f   __attribute__((ext_vector_type(8)));

#define DEVI __device__ __forceinline__

DEVI bf16_t f2bf(float f) { return (bf16_t)f; }  // native RNE cvt

DEVI float wave_sum(float v) {
  for (int o = 16; o; o >>= 1) v += __shfl_xor(v, o, 32);
  return v;
}
DEVI float wave_max(float v) {
  for (int o = 16; o; o >>= 1) v = fmaxf(v, __shfl_xor(v, o, 32));
  return v;
}
DEVI float sigf(float x) { return 1.0f / (1.0f + expf(-x)); }

// ---------------------------------------------------------------------------
// Batched strided GEMM:  C[bo,bi] (+)= alpha * A[bo,bi] @ B[bo,bi] (+bias)
//   A(m,k) at A + bo*sAb + bi*sAb2 + m*sAm + k*sAk   (fp32 -> bf16)
//   B(k,n) at B + bo*sBb + bi*sBb2 + k*sBk + n*sBn
//   C(m,n) at C + bo*sCb + bi*sCb2 + m*sCm + n*sCn
// ---------------------------------------------------------------------------
struct GemmP {
  const float *A, *B;
  float *C;
  const float *bias;
  int M, N, K, Bi;
  long long sAb, sAm, sAk, sAb2;
  long long sBb, sBk, sBn, sBb2;
  long long sCb, sCm, sCn, sCb2;
  float alpha;
  int accum;
};

// pack two contiguous 8-float runs (base[0..7], base[16..23]) into a bf16 frag
DEVI v16bf pack_frag_vec(const float* base) {
  const float4 x0 = *(const float4*)(base + 0);
  const float4 x1 = *(const float4*)(base + 4);
  const float4 y0 = *(const float4*)(base + 16);
  const float4 y1 = *(const float4*)(base + 20);
  v16bf f;
  f[0] = f2bf(x0.x);  f[1] = f2bf(x0.y);  f[2]  = f2bf(x0.z);  f[3]  = f2bf(x0.w);
  f[4] = f2bf(x1.x);  f[5] = f2bf(x1.y);  f[6]  = f2bf(x1.z);  f[7]  = f2bf(x1.w);
  f[8] = f2bf(y0.x);  f[9] = f2bf(y0.y);  f[10] = f2bf(y0.z);  f[11] = f2bf(y0.w);
  f[12] = f2bf(y1.x); f[13] = f2bf(y1.y); f[14] = f2bf(y1.z); f[15] = f2bf(y1.w);
  return f;
}

// branchless strided fragment load with K-tail zero masking
DEVI v16bf pack_frag_strided(const float* base, long long sK, int k0, int half, int K) {
  v16bf f;
#pragma unroll
  for (int j = 0; j < 16; ++j) {
    const int kk = k0 + ((j >> 3) << 4) + (half << 3) + (j & 7);
    const int kkc = (kk < K) ? kk : 0;           // clamp address (always valid)
    float v = base[(long long)kkc * sK];         // unconditional load
    f[j] = f2bf((kk < K) ? v : 0.0f);            // zero the K-pad
  }
  return f;
}

template <int AK1, int BK1>
__global__ __launch_bounds__(256) void k_gemm(GemmP p) {
  const int lane = threadIdx.x & 31;
  const int wv   = threadIdx.x >> 5;              // 8 waves: 2 (M) x 4 (N)
  const int tileM = (blockIdx.y * 2 + (wv >> 2)) * 32;   // wave owns 32x32 C
  const int tileN = (blockIdx.x * 4 + (wv & 3)) * 32;
  if (tileM >= p.M || tileN >= p.N) return;       // wave-uniform
  const long long bo = blockIdx.z / p.Bi;
  const long long bi = blockIdx.z % p.Bi;
  const float* A = p.A + bo * p.sAb + bi * p.sAb2;
  const float* B = p.B + bo * p.sBb + bi * p.sBb2;
  float*       C = p.C + bo * p.sCb + bi * p.sCb2;

  const int half = lane >> 4;
  const int l15  = lane & 15;
  const int am0 = tileM + l15,      am1 = tileM + 16 + l15;
  const int bn0 = tileN + l15,      bn1 = tileN + 16 + l15;
  // clamp OOB rows/cols to 0: their products only reach C entries that the
  // store masks out, so no value masking is needed.
  const float* Arow0 = A + (long long)((am0 < p.M) ? am0 : 0) * p.sAm;
  const float* Arow1 = A + (long long)((am1 < p.M) ? am1 : 0) * p.sAm;
  const float* Bcol0 = B + (long long)((bn0 < p.N) ? bn0 : 0) * p.sBn;
  const float* Bcol1 = B + (long long)((bn1 < p.N) ? bn1 : 0) * p.sBn;
  const int hofs = half << 3;

  v8f acc00 = {}, acc01 = {}, acc10 = {}, acc11 = {};
  const int kMain = (AK1 && BK1) ? (p.K & ~31) : ((AK1 || BK1) ? (p.K & ~31) : 0);
  int k0 = 0;
  for (; k0 < kMain; k0 += 32) {                 // hot loop: no branches
    v16bf af0, af1, bf0, bf1;
    if (AK1) { af0 = pack_frag_vec(Arow0 + k0 + hofs);
               af1 = pack_frag_vec(Arow1 + k0 + hofs); }
    else     { af0 = pack_frag_strided(Arow0, p.sAk, k0, half, p.K);
               af1 = pack_frag_strided(Arow1, p.sAk, k0, half, p.K); }
    if (BK1) { bf0 = pack_frag_vec(Bcol0 + k0 + hofs);
               bf1 = pack_frag_vec(Bcol1 + k0 + hofs); }
    else     { bf0 = pack_frag_strided(Bcol0, p.sBk, k0, half, p.K);
               bf1 = pack_frag_strided(Bcol1, p.sBk, k0, half, p.K); }
    acc00 = __builtin_amdgcn_wmma_f32_16x16x32_bf16(false, af0, false, bf0, (short)0, acc00, false, false);
    acc01 = __builtin_amdgcn_wmma_f32_16x16x32_bf16(false, af0, false, bf1, (short)0, acc01, false, false);
    acc10 = __builtin_amdgcn_wmma_f32_16x16x32_bf16(false, af1, false, bf0, (short)0, acc10, false, false);
    acc11 = __builtin_amdgcn_wmma_f32_16x16x32_bf16(false, af1, false, bf1, (short)0, acc11, false, false);
  }
  for (; k0 < p.K; k0 += 32) {                   // strided / K-tail path
    v16bf af0 = pack_frag_strided(Arow0, p.sAk, k0, half, p.K);
    v16bf af1 = pack_frag_strided(Arow1, p.sAk, k0, half, p.K);
    v16bf bf0 = pack_frag_strided(Bcol0, p.sBk, k0, half, p.K);
    v16bf bf1 = pack_frag_strided(Bcol1, p.sBk, k0, half, p.K);
    acc00 = __builtin_amdgcn_wmma_f32_16x16x32_bf16(false, af0, false, bf0, (short)0, acc00, false, false);
    acc01 = __builtin_amdgcn_wmma_f32_16x16x32_bf16(false, af0, false, bf1, (short)0, acc01, false, false);
    acc10 = __builtin_amdgcn_wmma_f32_16x16x32_bf16(false, af1, false, bf0, (short)0, acc10, false, false);
    acc11 = __builtin_amdgcn_wmma_f32_16x16x32_bf16(false, af1, false, bf1, (short)0, acc11, false, false);
  }

#pragma unroll
  for (int mi = 0; mi < 2; ++mi) {
    const v8f* accL = (mi == 0) ? &acc00 : &acc10;
    const v8f* accR = (mi == 0) ? &acc01 : &acc11;
#pragma unroll
    for (int r = 0; r < 8; ++r) {
      const int row = tileM + mi * 16 + r + half * 8;
      if (row < p.M) {
        const long long rb = (long long)row * p.sCm;
        const int col0 = tileN + l15;
        const int col1 = col0 + 16;
        if (col0 < p.N) {
          const long long ci = rb + (long long)col0 * p.sCn;
          float v = (*accL)[r] * p.alpha;
          if (p.bias)  v += p.bias[col0];
          if (p.accum) v += C[ci];
          C[ci] = v;
        }
        if (col1 < p.N) {
          const long long ci = rb + (long long)col1 * p.sCn;
          float v = (*accR)[r] * p.alpha;
          if (p.bias)  v += p.bias[col1];
          if (p.accum) v += C[ci];
          C[ci] = v;
        }
      }
    }
  }
}

// ---------------------------------------------------------------------------
// LayerNorm over last dim, one wave per row.
// ---------------------------------------------------------------------------
__global__ __launch_bounds__(256) void k_ln(const float* x, const float* g,
                                            const float* b, float* out,
                                            int rows, int D) {
  const int row  = blockIdx.x * 8 + (threadIdx.x >> 5);
  const int lane = threadIdx.x & 31;
  if (row >= rows) return;
  const float* xr = x + (long long)row * D;
  float s = 0.0f;
  for (int j = lane; j < D; j += 32) s += xr[j];
  const float mean = wave_sum(s) / (float)D;
  float v = 0.0f;
  for (int j = lane; j < D; j += 32) { float d = xr[j] - mean; v += d * d; }
  const float rstd = rsqrtf(wave_sum(v) / (float)D + 1e-5f);
  float* orow = out + (long long)row * D;
  for (int j = lane; j < D; j += 32)
    orow[j] = (xr[j] - mean) * rstd * g[j] + b[j];
}

// ---------------------------------------------------------------------------
// Softmax over K with additive bias[(q*K+k)*H+h] and INF*(mask-1) row mask.
// scores layout [bo, h, q, k] contiguous; one wave per (bo,h,q) row.
// ---------------------------------------------------------------------------
__global__ __launch_bounds__(256) void k_softmax(float* sc, const float* bias,
                                                 int H, const float* mask,
                                                 long long maskStride, int i0,
                                                 int Q, int K, int rows) {
  const int row  = blockIdx.x * 8 + (threadIdx.x >> 5);
  const int lane = threadIdx.x & 31;
  if (row >= rows) return;
  const int q  = row % Q;
  const int h  = (row / Q) % H;
  const int bo = row / (Q * H);
  float* r = sc + (long long)row * K;
  float vbuf[8];
  float m = -1e30f;
  int j = 0;
  for (int k = lane; k < K; k += 32) {
    float x = r[k] + bias[((long long)q * K + k) * H + h];
    if (mask) x += 1e9f * (mask[(long long)(i0 + bo) * maskStride + k] - 1.0f);
    vbuf[j++] = x;
    m = fmaxf(m, x);
  }
  m = wave_max(m);
  float s = 0.0f;
  j = 0;
  for (int k = lane; k < K; k += 32) { float e = expf(vbuf[j] - m); vbuf[j++] = e; s += e; }
  s = wave_sum(s);
  const float inv = 1.0f / s;
  j = 0;
  for (int k = lane; k < K; k += 32) r[k] = vbuf[j++] * inv;
}

// ---------------------------------------------------------------------------
// Elementwise kernels
// ---------------------------------------------------------------------------
__global__ void k_mul_sig(const float* a, const float* b, const float* rowmask,
                          int rowlen, float* out, long long n) {
  long long i = (long long)blockIdx.x * blockDim.x + threadIdx.x;
  if (i >= n) return;
  float v = a[i] * sigf(b[i]);
  if (rowmask) v *= rowmask[i / rowlen];
  out[i] = v;
}
__global__ void k_silu_mul(const float* a, const float* b, float* out, long long n) {
  long long i = (long long)blockIdx.x * blockDim.x + threadIdx.x;
  if (i >= n) return;
  float x = a[i];
  out[i] = x * sigf(x) * b[i];
}
__global__ void k_add(float* y, const float* x, long long n) {
  long long i = (long long)blockIdx.x * blockDim.x + threadIdx.x;
  if (i >= n) return;
  y[i] += x[i];
}
__global__ void k_add_T(float* z, const float* d, int n, int c) {
  long long i = (long long)blockIdx.x * blockDim.x + threadIdx.x;
  long long total = (long long)n * n * c;
  if (i >= total) return;
  int cc = (int)(i % c);
  int jj = (int)((i / c) % n);
  int ii = (int)(i / ((long long)c * n));
  z[i] += d[((long long)jj * n + ii) * c + cc];
}
__global__ void k_transpose(const float* in, float* out, int n, int c) {
  long long i = (long long)blockIdx.x * blockDim.x + threadIdx.x;
  long long total = (long long)n * n * c;
  if (i >= total) return;
  int cc = (int)(i % c);
  int jj = (int)((i / c) % n);
  int ii = (int)(i / ((long long)c * n));
  out[((long long)jj * n + ii) * c + cc] = in[i];
}

// ---------------------------------------------------------------------------
// Host-side helpers
// ---------------------------------------------------------------------------
static void gemm(hipStream_t st, const float* A, const float* B, float* C,
                 const float* bias, int M, int Nn, int K, float alpha, int accum,
                 long long sAm, long long sAk, long long sBk, long long sBn,
                 long long sCm, long long sCn,
                 int Bo = 1, int Bi = 1,
                 long long sAb = 0, long long sAb2 = 0,
                 long long sBb = 0, long long sBb2 = 0,
                 long long sCb = 0, long long sCb2 = 0) {
  GemmP p;
  p.A = A; p.B = B; p.C = C; p.bias = bias;
  p.M = M; p.N = Nn; p.K = K; p.Bi = Bi;
  p.sAb = sAb; p.sAm = sAm; p.sAk = sAk; p.sAb2 = sAb2;
  p.sBb = sBb; p.sBk = sBk; p.sBn = sBn; p.sBb2 = sBb2;
  p.sCb = sCb; p.sCm = sCm; p.sCn = sCn; p.sCb2 = sCb2;
  p.alpha = alpha; p.accum = accum;
  dim3 grid((Nn + 127) / 128, (M + 63) / 64, Bo * Bi);
  const bool ak1 = (sAk == 1), bk1 = (sBk == 1);
  if (ak1 && bk1)      k_gemm<1, 1><<<grid, 256, 0, st>>>(p);
  else if (ak1)        k_gemm<1, 0><<<grid, 256, 0, st>>>(p);
  else if (bk1)        k_gemm<0, 1><<<grid, 256, 0, st>>>(p);
  else                 k_gemm<0, 0><<<grid, 256, 0, st>>>(p);
}
static void lnorm(hipStream_t st, const float* x, const float* g, const float* b,
                  float* out, int rows, int D) {
  k_ln<<<(rows + 7) / 8, 256, 0, st>>>(x, g, b, out, rows, D);
}
static inline dim3 ewg(long long n) { return dim3((unsigned)((n + 255) / 256)); }

// ---------------------------------------------------------------------------
extern "C" void kernel_launch(void* const* d_in, const int* in_sizes, int n_in,
                              void* d_out, int out_size, void* d_ws, size_t ws_size,
                              hipStream_t stream) {
  (void)in_sizes; (void)n_in; (void)out_size; (void)ws_size;
  const int N = 192, S = 384, Z = 128, HS = 16, HP = 4, CP = 32;
  const int NN = N * N;
  const long long NNZ = (long long)NN * Z;

  float* s = (float*)d_out;                 // [N,S]
  float* z = s + (long long)N * S;          // [N,N,Z]
  hipMemcpyAsync(s, d_in[0], sizeof(float) * (size_t)N * S,
                 hipMemcpyDeviceToDevice, stream);
  hipMemcpyAsync(z, d_in[1], sizeof(float) * (size_t)NNZ,
                 hipMemcpyDeviceToDevice, stream);
  const float* mask  = (const float*)d_in[2];   // [N]
  const float* pmask = (const float*)d_in[3];   // [N,N]

  // ---- workspace carve-up (floats) ----
  float* wsf = (float*)d_ws;
  long long off = 0;
  auto alloc = [&](long long nf) { float* p = wsf + off; off += nf; return p; };
  float* xln    = alloc(NNZ);                  // ln(z)
  float* bufP   = alloc((long long)NN * 256);  // also q|k during attention
  float* bufG   = alloc((long long)NN * 256);  // also v|g during attention
  float* bufT   = alloc(NNZ);                  // tri product / attn output o
  float* bufTln = alloc(NNZ);                  // ln(t) / gated o
  float* bufD   = alloc(NNZ);                  // delta (also zT staging)
  float* pmT    = alloc(NN);                   // transposed pair mask
  float* tbb    = alloc((long long)NN * HP);   // triangle bias
  float* scc    = alloc((long long)16 * HP * N * N);  // score chunk
  float* sn = alloc((long long)N * S);
  float* sq = alloc((long long)N * S);
  float* sk = alloc((long long)N * S);
  float* sv = alloc((long long)N * S);
  float* sg = alloc((long long)N * S);
  float* so = alloc((long long)N * S);
  float* zbb = alloc((long long)NN * HS);

  for (int L = 0; L < 2; ++L) {
    auto pp = [&](int j) { return (const float*)d_in[4 + L * 53 + j]; };

    // ================= triangle multiplication (outgoing, incoming) ========
    for (int t = 0; t < 2; ++t) {
      const int b0 = 2 + t * 8;
      const bool outgoing = (t == 0);
      lnorm(stream, z, pp(b0 + 0), pp(b0 + 1), xln, NN, Z);
      gemm(stream, xln, pp(b0 + 2), bufP, nullptr, NN, 2 * Z, Z, 1.f, 0,
           Z, 1, 2 * Z, 1, 2 * Z, 1);
      gemm(stream, xln, pp(b0 + 3), bufG, nullptr, NN, 2 * Z, Z, 1.f, 0,
           Z, 1, 2 * Z, 1, 2 * Z, 1);
      { long long n = (long long)NN * 2 * Z;
        k_mul_sig<<<ewg(n), 256, 0, stream>>>(bufP, bufG, pmask, 2 * Z, bufP, n); }
      // triangle einsum: batch over d (128), a = bufP[:, :Z], b = bufP[:, Z:]
      if (outgoing)
        gemm(stream, bufP, bufP + Z, bufT, nullptr, N, N, N, 1.f, 0,
             (long long)N * 2 * Z, 2 * Z, 2 * Z, (long long)N * 2 * Z,
             (long long)N * Z, Z, /*Bo=*/Z, 1, /*sAb=*/1, 0, /*sBb=*/1, 0, /*sCb=*/1, 0);
      else
        gemm(stream, bufP, bufP + Z, bufT, nullptr, N, N, N, 1.f, 0,
             2 * Z, (long long)N * 2 * Z, (long long)N * 2 * Z, 2 * Z,
             (long long)N * Z, Z, Z, 1, 1, 0, 1, 0, 1, 0);
      lnorm(stream, bufT, pp(b0 + 4), pp(b0 + 5), bufTln, NN, Z);
      gemm(stream, bufTln, pp(b0 + 6), bufD, nullptr, NN, Z, Z, 1.f, 0,
           Z, 1, Z, 1, Z, 1);
      gemm(stream, xln, pp(b0 + 7), bufG, nullptr, NN, Z, Z, 1.f, 0,
           Z, 1, Z, 1, Z, 1);
      k_mul_sig<<<ewg(NNZ), 256, 0, stream>>>(bufD, bufG, nullptr, 1, bufD, NNZ);
      k_add<<<ewg(NNZ), 256, 0, stream>>>(z, bufD, NNZ);
    }

    // ================= triangle attention (start, end) =====================
    for (int t = 0; t < 2; ++t) {
      const int b0 = 18 + t * 8;
      const bool endv = (t == 1);
      const float* zin = z;
      const float* pm  = pmask;
      if (endv) {
        k_transpose<<<ewg(NNZ), 256, 0, stream>>>(z, bufD, N, Z);
        k_transpose<<<ewg(NN), 256, 0, stream>>>(pmask, pmT, N, 1);
        zin = bufD; pm = pmT;
      }
      lnorm(stream, zin, pp(b0 + 0), pp(b0 + 1), xln, NN, Z);
      float* q = bufP;
      float* k = bufP + NNZ;
      float* v = bufG;
      float* g = bufG + NNZ;
      gemm(stream, xln, pp(b0 + 3), q, nullptr, NN, Z, Z, 1.f, 0, Z, 1, Z, 1, Z, 1);
      gemm(stream, xln, pp(b0 + 4), k, nullptr, NN, Z, Z, 1.f, 0, Z, 1, Z, 1, Z, 1);
      gemm(stream, xln, pp(b0 + 5), v, nullptr, NN, Z, Z, 1.f, 0, Z, 1, Z, 1, Z, 1);
      gemm(stream, xln, pp(b0 + 2), tbb, nullptr, NN, HP, Z, 1.f, 0, Z, 1, HP, 1, HP, 1);
      for (int i0 = 0; i0 < N; i0 += 16) {
        // scores[bo,h,q,k] = (q . k)/sqrt(CP)
        gemm(stream, q + (long long)i0 * N * Z, k + (long long)i0 * N * Z, scc,
             nullptr, N, N, CP, 1.0f / sqrtf((float)CP), 0,
             Z, 1, 1, Z, N, 1,
             /*Bo=*/16, /*Bi=*/HP,
             (long long)N * Z, CP, (long long)N * Z, CP,
             (long long)HP * N * N, (long long)N * N);
        const int rows = 16 * HP * N;
        k_softmax<<<(rows + 7) / 8, 256, 0, stream>>>(
            scc, tbb, HP, pm, (long long)N, i0, N, N, rows);
        // o[i,q,h,c] = probs @ v
        gemm(stream, scc, v + (long long)i0 * N * Z, bufT + (long long)i0 * N * Z,
             nullptr, N, CP, N, 1.f, 0,
             N, 1, Z, 1, Z, 1,
             16, HP,
             (long long)HP * N * N, (long long)N * N,
             (long long)N * Z, CP,
             (long long)N * Z, CP);
      }
      gemm(stream, xln, pp(b0 + 6), g, nullptr, NN, Z, Z, 1.f, 0, Z, 1, Z, 1, Z, 1);
      k_mul_sig<<<ewg(NNZ), 256, 0, stream>>>(bufT, g, nullptr, 1, bufTln, NNZ);
      gemm(stream, bufTln, pp(b0 + 7), bufD, nullptr, NN, Z, Z, 1.f, 0,
           Z, 1, Z, 1, Z, 1);
      if (!endv) k_add<<<ewg(NNZ), 256, 0, stream>>>(z, bufD, NNZ);
      else       k_add_T<<<ewg(NNZ), 256, 0, stream>>>(z, bufD, N, Z);
    }

    // ================= pair transition (tz, leaves 48..52) =================
    lnorm(stream, z, pp(48), pp(49), xln, NN, Z);
    {
      const int RC = 4608;
      for (int r0 = 0; r0 < NN; r0 += RC) {
        gemm(stream, xln + (long long)r0 * Z, pp(50), bufP, nullptr,
             RC, 4 * Z, Z, 1.f, 0, Z, 1, 4 * Z, 1, 4 * Z, 1);
        gemm(stream, xln + (long long)r0 * Z, pp(51), bufG, nullptr,
             RC, 4 * Z, Z, 1.f, 0, Z, 1, 4 * Z, 1, 4 * Z, 1);
        long long n = (long long)RC * 4 * Z;
        k_silu_mul<<<ewg(n), 256, 0, stream>>>(bufP, bufG, bufP, n);
        gemm(stream, bufP, pp(52), z + (long long)r0 * Z, nullptr,
             RC, Z, 4 * Z, 1.f, /*accum=*/1, 4 * Z, 1, Z, 1, Z, 1);
      }
    }

    // ================= attention with pair bias (apb, leaves 34..42) =======
    {
      const int d = S / HS;  // 24
      lnorm(stream, s, pp(0), pp(1), sn, N, S);
      gemm(stream, sn, pp(34), sq, pp(35), N, S, S, 1.f, 0, S, 1, S, 1, S, 1);
      gemm(stream, sn, pp(36), sk, nullptr, N, S, S, 1.f, 0, S, 1, S, 1, S, 1);
      gemm(stream, sn, pp(37), sv, nullptr, N, S, S, 1.f, 0, S, 1, S, 1, S, 1);
      gemm(stream, sn, pp(38), sg, nullptr, N, S, S, 1.f, 0, S, 1, S, 1, S, 1);
      lnorm(stream, z, pp(39), pp(40), xln, NN, Z);
      gemm(stream, xln, pp(41), zbb, nullptr, NN, HS, Z, 1.f, 0, Z, 1, HS, 1, HS, 1);
      // scores[h,q,k], K=24 zero-padded to 32 by tail masking
      gemm(stream, sq, sk, scc, nullptr, N, N, d, 1.0f / sqrtf((float)d), 0,
           S, 1, 1, S, N, 1,
           /*Bo=*/HS, 1, d, 0, d, 0, (long long)N * N, 0);
      const int rows = HS * N;
      k_softmax<<<(rows + 7) / 8, 256, 0, stream>>>(
          scc, zbb, HS, mask, 0, 0, N, N, rows);
      gemm(stream, scc, sv, so, nullptr, N, d, N, 1.f, 0,
           N, 1, S, 1, S, 1,
           HS, 1, (long long)N * N, 0, d, 0, d, 0);
      long long n = (long long)N * S;
      k_mul_sig<<<ewg(n), 256, 0, stream>>>(so, sg, nullptr, 1, so, n);
      gemm(stream, so, pp(42), s, nullptr, N, S, S, 1.f, /*accum=*/1,
           S, 1, S, 1, S, 1);
    }

    // ================= single transition (ts, leaves 43..47) ===============
    {
      lnorm(stream, s, pp(43), pp(44), sn, N, S);
      gemm(stream, sn, pp(45), bufP, nullptr, N, 4 * S, S, 1.f, 0,
           S, 1, 4 * S, 1, 4 * S, 1);
      gemm(stream, sn, pp(46), bufG, nullptr, N, 4 * S, S, 1.f, 0,
           S, 1, 4 * S, 1, 4 * S, 1);
      long long n = (long long)N * 4 * S;
      k_silu_mul<<<ewg(n), 256, 0, stream>>>(bufP, bufG, bufP, n);
      gemm(stream, bufP, pp(47), s, nullptr, N, S, 4 * S, 1.f, /*accum=*/1,
           4 * S, 1, S, 1, S, 1);
    }
  }
}
